// IntegratedQuanvolutionTransformer_65481071405816
// MI455X (gfx1250) — matrix operand order repaired and checked by reading, not compiled, not run on hardware
//
#include <hip/hip_runtime.h>
#include <hip/hip_bf16.h>

typedef _Float16 half_t;
typedef __attribute__((ext_vector_type(16))) _Float16 v16h;
typedef __attribute__((ext_vector_type(8)))  float    v8f;

#define B_   256
#define S_   196
#define E_   256
#define H_   8
#define NB_  4
#define F_   1024
#define NQ_  8
#define NC_  10
#define NTOK (B_ * S_)   // 50176, divisible by 64

// ---------------------------------------------------------------------------
// WMMA helpers: per-lane LDS gathers in a contiguous k order used by both A and B.
// A (16x32 f16): lane L -> row m = L%16; VGPR v holds K pair starting at
//   k0 = 2*v + 16*(L/16)
// B (32x16 f16): lane L -> col n = L%16; VGPR v holds K pair starting at
//   k0 = 2*v + 16*(L/16)
// C/D (16x16 f32): VGPR r, lane L -> row m = r + 8*(L/16), col n = L%16
// ---------------------------------------------------------------------------
__device__ __forceinline__ v16h lds_load_A(const half_t* p, int lane, int lda) {
  int m = lane & 15, kh = lane >> 4;
  v16h a;
#pragma unroll
  for (int v = 0; v < 8; ++v) {
    int k0 = 2 * v + 16 * kh;
    a[2 * v]     = p[m * lda + k0];
    a[2 * v + 1] = p[m * lda + k0 + 1];
  }
  return a;
}

__device__ __forceinline__ v16h lds_load_B(const half_t* p, int lane, int ldb) {
  int n = lane & 15, kh = lane >> 4;
  v16h b;
#pragma unroll
  for (int v = 0; v < 8; ++v) {
    int k0 = 2 * v + 16 * kh;
    b[2 * v]     = p[n * ldb + k0];
    b[2 * v + 1] = p[n * ldb + k0 + 1];
  }
  return b;
}

__device__ __forceinline__ v8f wmma_f16(v16h a, v16h b, v8f c) {
  return __builtin_amdgcn_wmma_f32_16x16x32_f16(false, a, false, b, (short)0, c,
                                                false, false);
}

// ---------------------------------------------------------------------------
// f32 -> f16 weight conversion (one-time)
// ---------------------------------------------------------------------------
__global__ void k_f32_to_f16(const float* __restrict__ src,
                             half_t* __restrict__ dst, int n) {
  int i = blockIdx.x * blockDim.x + threadIdx.x;
  if (i < n) dst[i] = (half_t)src[i];
}

// ---------------------------------------------------------------------------
// Quanvolution (depthwise 2x2 s2 + pointwise 1x1) + channel-major .view
// + embed (K=4) + sinusoidal positional encoding.  One block per token.
// ---------------------------------------------------------------------------
__global__ void k_quanv_embed(const float* __restrict__ x,
                              const float* __restrict__ qfw,
                              const float* __restrict__ qfb,
                              const float* __restrict__ pww,
                              const float* __restrict__ pwb,
                              const float* __restrict__ ew,
                              const float* __restrict__ eb,
                              float* __restrict__ h) {
  int tok = blockIdx.x;          // 0..50175
  int b = tok / S_, s = tok % S_;
  __shared__ float sfeat[4];
  int t = threadIdx.x;
  if (t < 4) {
    // flat index into the [4,14,14] channel-major conv output
    int i = s * 4 + t;
    int co = i / S_, p = i % S_;
    int oh = p / 14, ow = p % 14;
    float acc = pwb[co];
#pragma unroll
    for (int ci = 0; ci < 4; ++ci) {
      float dw = qfb[ci];
#pragma unroll
      for (int kh = 0; kh < 2; ++kh)
#pragma unroll
        for (int kw = 0; kw < 2; ++kw)
          dw += x[(((size_t)b * 4 + ci) * 28 + (2 * oh + kh)) * 28 +
                  (2 * ow + kw)] * qfw[ci * 4 + kh * 2 + kw];
      acc += dw * pww[co * 4 + ci];
    }
    sfeat[t] = acc;
  }
  __syncthreads();
  int e = t;
  float v = eb[e];
#pragma unroll
  for (int j = 0; j < 4; ++j) v += sfeat[j] * ew[e * 4 + j];
  int k = e >> 1;
  float ang = (float)s * __expf((float)(2 * k) * (-9.210340371976184f / 256.0f));
  v += (e & 1) ? __cosf(ang) : __sinf(ang);
  h[(size_t)tok * E_ + e] = v;
}

// ---------------------------------------------------------------------------
// C[M,N] = A[M,K] @ W[N,K]^T + bias   (A f32 or f16, W f16, C f32 or f16)
// 64x64 block tile, BK=32, 8 waves: wave (wm,wn) owns 16 rows x 32 cols.
// Requires M%64==0, N%64==0, K%32==0.  Prefetches next k-tile (CDNA5
// global_prefetch path).
// ---------------------------------------------------------------------------
template <typename AT, typename OT>
__global__ void k_gemm_bias(const AT* __restrict__ A,
                            const half_t* __restrict__ W,
                            const float* __restrict__ bias,
                            OT* __restrict__ C, int M, int N, int K) {
  __shared__ half_t As[64 * 34];
  __shared__ half_t Ws[64 * 34];
  int tid = threadIdx.x;
  int lane = tid & 31, wave = tid >> 5;
  int wm = wave & 3, wn = wave >> 2;
  int m0 = blockIdx.y * 64, n0 = blockIdx.x * 64;
  v8f c0 = {}, c1 = {};
  for (int kk = 0; kk < K; kk += 32) {
    __syncthreads();
#pragma unroll
    for (int it = 0; it < 8; ++it) {
      int idx = tid + it * 256;
      int r = idx >> 5, cx = idx & 31;
      As[r * 34 + cx] = (half_t)A[(size_t)(m0 + r) * K + kk + cx];
      Ws[r * 34 + cx] = W[(size_t)(n0 + r) * K + kk + cx];
    }
    if (kk + 32 < K) {
      // prefetch one row (8*wave) of the next k-tile of A and W
      __builtin_prefetch(&A[(size_t)(m0 + (tid >> 5) * 8) * K + kk + 32], 0, 3);
      __builtin_prefetch(&W[(size_t)(n0 + (tid >> 5) * 8) * K + kk + 32], 0, 3);
    }
    __syncthreads();
    v16h a  = lds_load_A(&As[wm * 16 * 34], lane, 34);
    v16h b0 = lds_load_B(&Ws[(wn * 32) * 34], lane, 34);
    v16h b1 = lds_load_B(&Ws[(wn * 32 + 16) * 34], lane, 34);
    c0 = wmma_f16(a, b0, c0);
    c1 = wmma_f16(a, b1, c1);
  }
  int mbase = m0 + wm * 16 + (lane >> 4) * 8;
  int ncol  = n0 + wn * 32 + (lane & 15);
#pragma unroll
  for (int r = 0; r < 8; ++r) {
    int row = mbase + r;
    C[(size_t)row * N + ncol]      = (OT)(c0[r] + bias[ncol]);
    C[(size_t)row * N + ncol + 16] = (OT)(c1[r] + bias[ncol + 16]);
  }
}

// ---------------------------------------------------------------------------
// Flash attention, one block per (b, head). 13 waves, one q-tile (16 rows)
// per wave; K/V staged once in LDS (V transposed for the B layout).
// hd=32 == WMMA K, so QK^T is a single wmma per 16-key tile.
// qkv is f16 (written by the QKV GEMM); output written f16.
// ---------------------------------------------------------------------------
__global__ void k_attn(const half_t* __restrict__ qkv, half_t* __restrict__ obuf) {
  int bh = blockIdx.x;
  int b = bh >> 3, hh = bh & 7;
  __shared__ half_t Ks[224 * 32];      // [key][dim]
  __shared__ half_t Vt[32 * 224];      // [dim][key]
  __shared__ half_t Ps[13 * 16 * 32];  // per-wave P staging
  int tid = threadIdx.x;
  const float scale = 0.17677669529663687f;  // 1/sqrt(32)
  const size_t baseb = (size_t)b * S_ * 768;

  for (int idx = tid; idx < 224 * 32; idx += 416) {
    int s = idx >> 5, d = idx & 31;
    half_t kv = (half_t)0, vv = (half_t)0;
    if (s < S_) {
      kv = qkv[baseb + (size_t)s * 768 + 256 + hh * 32 + d];
      vv = qkv[baseb + (size_t)s * 768 + 512 + hh * 32 + d];
    }
    Ks[s * 32 + d]  = kv;
    Vt[d * 224 + s] = vv;
  }
  __syncthreads();

  int wave = tid >> 5, lane = tid & 31;
  int q0 = wave * 16;
  int m = lane & 15, kh = lane >> 4;
  int qs = q0 + m;
  v16h aq;
#pragma unroll
  for (int v = 0; v < 8; ++v) {
    int k0 = 2 * v + 16 * kh;
#pragma unroll
    for (int j = 0; j < 2; ++j) {
      aq[2 * v + j] = (qs < S_)
          ? qkv[baseb + (size_t)qs * 768 + hh * 32 + k0 + j]
          : (half_t)0;
    }
  }

  float mrow[8], lrow[8];
  v8f o0 = {}, o1 = {};
#pragma unroll
  for (int r = 0; r < 8; ++r) { mrow[r] = -1e30f; lrow[r] = 0.f; }
  half_t* Pw = &Ps[wave * 16 * 32];

  for (int ch = 0; ch < 7; ++ch) {           // 7 chunks of 32 keys (224 padded)
    v16h bk0 = lds_load_B(&Ks[(ch * 32) * 32], lane, 32);
    v16h bk1 = lds_load_B(&Ks[(ch * 32 + 16) * 32], lane, 32);
    v8f z = {};
    v8f s0 = wmma_f16(aq, bk0, z);
    v8f s1 = wmma_f16(aq, bk1, z);
    int key0 = ch * 32 + (lane & 15);
#pragma unroll
    for (int r = 0; r < 8; ++r) {
      s0[r] *= scale;
      s1[r] *= scale;
    }
    if (key0 >= S_) {
#pragma unroll
      for (int r = 0; r < 8; ++r) s0[r] = -1e30f;
    }
    if (key0 + 16 >= S_) {
#pragma unroll
      for (int r = 0; r < 8; ++r) s1[r] = -1e30f;
    }
#pragma unroll
    for (int r = 0; r < 8; ++r) {
      float vmax = fmaxf(s0[r], s1[r]);
#pragma unroll
      for (int mk = 1; mk < 16; mk <<= 1)
        vmax = fmaxf(vmax, __shfl_xor(vmax, mk, 32));
      float mn = fmaxf(mrow[r], vmax);
      float alpha = __expf(mrow[r] - mn);
      float p0 = __expf(s0[r] - mn);
      float p1 = __expf(s1[r] - mn);
      float rs = p0 + p1;
#pragma unroll
      for (int mk = 1; mk < 16; mk <<= 1) rs += __shfl_xor(rs, mk, 32);
      lrow[r] = lrow[r] * alpha + rs;
      mrow[r] = mn;
      o0[r] *= alpha;
      o1[r] *= alpha;
      Pw[(r + 8 * kh) * 32 + (lane & 15)]      = (half_t)p0;
      Pw[(r + 8 * kh) * 32 + 16 + (lane & 15)] = (half_t)p1;
    }
    // wave-private LDS round trip: CDNA5 split-counter wait on DS ops
    asm volatile("s_wait_dscnt 0" ::: "memory");
    v16h ap  = lds_load_A(Pw, lane, 32);
    v16h bv0 = lds_load_B(&Vt[ch * 32], lane, 224);
    v16h bv1 = lds_load_B(&Vt[16 * 224 + ch * 32], lane, 224);
    o0 = wmma_f16(ap, bv0, o0);
    o1 = wmma_f16(ap, bv1, o1);
  }

#pragma unroll
  for (int r = 0; r < 8; ++r) {
    int row = q0 + r + 8 * kh;
    if (row < S_) {
      float inv = 1.0f / lrow[r];
      size_t o = ((size_t)b * S_ + row) * E_ + hh * 32 + (lane & 15);
      obuf[o]      = (half_t)(o0[r] * inv);
      obuf[o + 16] = (half_t)(o1[r] * inv);
    }
  }
}

// ---------------------------------------------------------------------------
// Quantum FFN, fused: qm=cos(h[:8])*cos(theta) -> l1(K=8)+ReLU in LDS (f16)
// -> l2 GEMM (K=1024) via WMMA with B gathered from pre-converted f16 weights.
// One block per 16 tokens; 8 waves each own 32 output columns.
// ---------------------------------------------------------------------------
__global__ void k_ffn(const float* __restrict__ h, const float* __restrict__ theta,
                      const float* __restrict__ l1w, const float* __restrict__ l1b,
                      const half_t* __restrict__ l2w, const float* __restrict__ l2b,
                      float* __restrict__ out) {
  __shared__ float qs[16 * 8];
  __shared__ half_t Fs[16 * 1032];
  int blk = blockIdx.x;
  int tid = threadIdx.x;
  if (tid < 128) {
    int tok = tid >> 3, j = tid & 7;
    int gt = blk * 16 + tok;
    qs[tok * 8 + j] = __cosf(h[(size_t)gt * E_ + j]) * __cosf(theta[j]);
  }
  __syncthreads();
  for (int idx = tid; idx < 16 * F_; idx += 256) {
    int tok = idx & 15, f = idx >> 4;
    float acc = l1b[f];
#pragma unroll
    for (int j = 0; j < 8; ++j) acc += qs[tok * 8 + j] * l1w[f * 8 + j];
    Fs[tok * 1032 + f] = (half_t)fmaxf(acc, 0.f);
  }
  __syncthreads();
  int lane = tid & 31, wave = tid >> 5;
  int n0 = wave * 32;
  int n = lane & 15, khh = lane >> 4;
  v8f c0 = {}, c1 = {};
  for (int kc = 0; kc < F_; kc += 32) {
    v16h a = lds_load_A(Fs + kc, lane, 1032);
    v16h b0, b1;
#pragma unroll
    for (int v = 0; v < 8; ++v) {
      int k0 = 2 * v + 16 * khh;
      b0[2 * v]     = l2w[(size_t)(n0 + n) * F_ + kc + k0];
      b0[2 * v + 1] = l2w[(size_t)(n0 + n) * F_ + kc + k0 + 1];
      b1[2 * v]     = l2w[(size_t)(n0 + 16 + n) * F_ + kc + k0];
      b1[2 * v + 1] = l2w[(size_t)(n0 + 16 + n) * F_ + kc + k0 + 1];
    }
    c0 = wmma_f16(a, b0, c0);
    c1 = wmma_f16(a, b1, c1);
  }
  int mbase = (lane >> 4) * 8;
#pragma unroll
  for (int r = 0; r < 8; ++r) {
    int tok = blk * 16 + mbase + r;
    int col = n0 + (lane & 15);
    out[(size_t)tok * E_ + col]      = c0[r] + l2b[col];
    out[(size_t)tok * E_ + col + 16] = c1[r] + l2b[col + 16];
  }
}

// ---------------------------------------------------------------------------
// Residual + LayerNorm, one block (256 threads) per row.
// ---------------------------------------------------------------------------
__device__ __forceinline__ float block_sum256(float v, float* red) {
#pragma unroll
  for (int mk = 16; mk >= 1; mk >>= 1) v += __shfl_xor(v, mk, 32);
  int wave = threadIdx.x >> 5, lane = threadIdx.x & 31;
  if (lane == 0) red[wave] = v;
  __syncthreads();
  if (wave == 0) {
    float t = (lane < 8) ? red[lane] : 0.f;
#pragma unroll
    for (int mk = 4; mk >= 1; mk >>= 1) t += __shfl_xor(t, mk, 32);
    if (lane == 0) red[8] = t;
  }
  __syncthreads();
  float o = red[8];
  __syncthreads();
  return o;
}

__global__ void k_resln(float* __restrict__ h, const float* __restrict__ d,
                        const float* __restrict__ g, const float* __restrict__ bt) {
  __shared__ float red[9];
  int row = blockIdx.x, e = threadIdx.x;
  size_t idx = (size_t)row * E_ + e;
  float v = h[idx] + d[idx];
  float mean = block_sum256(v, red) * (1.0f / E_);
  float dv = v - mean;
  float var = block_sum256(dv * dv, red) * (1.0f / E_);
  h[idx] = dv * rsqrtf(var + 1e-5f) * g[e] + bt[e];
}

// ---------------------------------------------------------------------------
// Mean-pool over S + classifier (E=256 -> NC=10). One block per batch item.
// ---------------------------------------------------------------------------
__global__ void k_pool_cls(const float* __restrict__ h, const float* __restrict__ cw,
                           const float* __restrict__ cb, float* __restrict__ out) {
  __shared__ float pool[E_];
  int b = blockIdx.x, e = threadIdx.x;
  float acc = 0.f;
  for (int s = 0; s < S_; ++s) acc += h[((size_t)b * S_ + s) * E_ + e];
  pool[e] = acc * (1.0f / S_);
  __syncthreads();
  if (e < NC_) {
    float o = cb[e];
    for (int k = 0; k < E_; ++k) o += pool[k] * cw[e * E_ + k];
    out[b * NC_ + e] = o;
  }
}

// ---------------------------------------------------------------------------
extern "C" void kernel_launch(void* const* d_in, const int* in_sizes, int n_in,
                              void* d_out, int out_size, void* d_ws, size_t ws_size,
                              hipStream_t stream) {
  const float* x     = (const float*)d_in[0];
  const float* qfw   = (const float*)d_in[1];
  const float* qfb   = (const float*)d_in[2];
  const float* pww   = (const float*)d_in[3];
  const float* pwb   = (const float*)d_in[4];
  const float* ew    = (const float*)d_in[5];
  const float* eb    = (const float*)d_in[6];
  const float* ipw   = (const float*)d_in[7];
  const float* ipb   = (const float*)d_in[8];
  const float* ow    = (const float*)d_in[9];
  const float* ob    = (const float*)d_in[10];
  const float* theta = (const float*)d_in[11];
  const float* l1w   = (const float*)d_in[12];
  const float* l1b   = (const float*)d_in[13];
  const float* l2w   = (const float*)d_in[14];
  const float* l2b   = (const float*)d_in[15];
  const float* ln1g  = (const float*)d_in[16];
  const float* ln1b  = (const float*)d_in[17];
  const float* ln2g  = (const float*)d_in[18];
  const float* ln2b  = (const float*)d_in[19];
  const float* cw    = (const float*)d_in[20];
  const float* cb    = (const float*)d_in[21];
  float* out = (float*)d_out;

  // workspace layout (sequential kernel ordering makes the qkv/proj alias safe:
  // attention fully consumes qkv16 before the out-proj GEMM writes proj)
  char* wsp = (char*)d_ws;
  float*  h      = (float*)wsp;                    wsp += (size_t)NTOK * E_ * 4;
  char*   qkvreg = wsp;                            wsp += (size_t)NTOK * 768 * 2;
  half_t* qkv16  = (half_t*)qkvreg;                // [NTOK, 768] f16
  float*  proj   = (float*)qkvreg;                 // [NTOK, 256] f32 (alias)
  half_t* attn16 = (half_t*)wsp;                   wsp += (size_t)NTOK * E_ * 2;
  float*  ffnout = (float*)wsp;                    wsp += (size_t)NTOK * E_ * 4;
  half_t* ipw16  = (half_t*)wsp;                   wsp += (size_t)NB_ * 768 * 256 * 2;
  half_t* ow16   = (half_t*)wsp;                   wsp += (size_t)NB_ * 256 * 256 * 2;
  half_t* l2w16  = (half_t*)wsp;

  k_f32_to_f16<<<(NB_ * 768 * 256 + 255) / 256, 256, 0, stream>>>(ipw, ipw16, NB_ * 768 * 256);
  k_f32_to_f16<<<(NB_ * 256 * 256 + 255) / 256, 256, 0, stream>>>(ow, ow16, NB_ * 256 * 256);
  k_f32_to_f16<<<(NB_ * 256 * 1024 + 255) / 256, 256, 0, stream>>>(l2w, l2w16, NB_ * 256 * 1024);

  k_quanv_embed<<<NTOK, 256, 0, stream>>>(x, qfw, qfb, pww, pwb, ew, eb, h);

  for (int i = 0; i < NB_; ++i) {
    k_gemm_bias<float, half_t><<<dim3(768 / 64, NTOK / 64), 256, 0, stream>>>(
        h, ipw16 + (size_t)i * 768 * 256, ipb + i * 768, qkv16, NTOK, 768, 256);
    k_attn<<<B_ * H_, 416, 0, stream>>>(qkv16, attn16);
    k_gemm_bias<half_t, float><<<dim3(256 / 64, NTOK / 64), 256, 0, stream>>>(
        attn16, ow16 + (size_t)i * 256 * 256, ob + i * 256, proj, NTOK, 256, 256);
    k_resln<<<NTOK, 256, 0, stream>>>(h, proj, ln1g + i * 256, ln1b + i * 256);
    k_ffn<<<NTOK / 16, 256, 0, stream>>>(h, theta + i * 8, l1w + i * 1024 * 8,
                                         l1b + i * 1024,
                                         l2w16 + (size_t)i * 256 * 1024,
                                         l2b + i * 256, ffnout);
    k_resln<<<NTOK, 256, 0, stream>>>(h, ffnout, ln2g + i * 256, ln2b + i * 256);
  }
  k_pool_cls<<<B_, 256, 0, stream>>>(h, cw, cb, out);
}
